// DAGERC_fushion_46548855554125
// MI455X (gfx1250) — matrix operand order repaired and checked by reading, not compiled, not run on hardware
//
#include <hip/hip_runtime.h>
#include <cstddef>

// ---------------- model dims ----------------
#define BB    64      // batch
#define NN    128     // sequence / nodes
#define HIDD  512
#define EMBD  1024
#define GDIM  1536    // 3*HID (GRU gates)
#define LDIM  2048    // 4*HID (LSTM gates)
#define INDIM 2560    // HID*3 + EMB
#define ROWS  8192    // BB*NN
#define NCLS  6

typedef __bf16 bf16_t;
typedef __bf16 v16bf __attribute__((ext_vector_type(16)));
typedef float  v8f   __attribute__((ext_vector_type(8)));

__device__ __forceinline__ float sigmoidf_(float x) { return 1.0f / (1.0f + expf(-x)); }

// K-permutation (swap bits 3<->4): within each 32 block, order [0-7,16-23,8-15,24-31].
// Applied to every bf16 activation buffer's column index at write time. GEMM sums over
// K, so permuting A's K-columns is exact as long as the hardware's assumed fragment
// mapping equals the actual logical k -- this permutation makes a lane's 16 A-values
// one contiguous 32B run at offset kh*16 while leaving the B fragment mapping unchanged.
__device__ __forceinline__ int swz(int c)
{
    return (c & ~24) | ((c & 8) << 1) | ((c & 16) >> 1);
}

#define WMMA_BF16(a, b, c) \
    __builtin_amdgcn_wmma_f32_16x16x32_bf16(false, a, false, b, (short)0, c, false, false)

// ---------------------------------------------------------------------------
// WMMA GEMM:  C[M,N] = A_bf16[M,K](K-swizzled) @ Wt_bf16[K,N] (+bias)(+bias2)(relu)
// One wave32 computes a 16x32 output (two 16x16 tiles sharing the A fragment).
// A fragments are direct v16bf loads (2 adjacent B128s); B is gathered per element at stride N; the K-loop is an
// explicit ping-pong pipeline (K % 64 == 0 for every shape used here), so
// WMMAs on one buffer overlap the in-flight loads of the other.
// A frag: lane r=lane&15 -> row m0+r; contiguous 16 elems at K = k0 + kh*16.
// B frag: lane holds col n0+(lane&15) (b1: +16); elem s -> K = k0 + kh*8 + s + (s&8).
// C/D: VGPR v -> M = m0 + kh*8 + v ; N = n0 + (lane&15).
// ---------------------------------------------------------------------------
__global__ void gemm_bf16(const bf16_t* __restrict__ A, const bf16_t* __restrict__ Wt,
                          const float* __restrict__ bias, const float* __restrict__ bias2,
                          float* __restrict__ C, bf16_t* __restrict__ Cbf,
                          int M, int N, int K, int act)
{
    const int lane   = threadIdx.x & 31;
    const int wave   = threadIdx.x >> 5;
    const int tilesN2 = N >> 5;               // 16x32 macro-tiles along N
    const int tilesM  = M >> 4;
    const int tile = blockIdx.x * (blockDim.x >> 5) + wave;
    if (tile >= tilesM * tilesN2) return;
    const int tm = tile / tilesN2, tn = tile % tilesN2;
    const int m0 = tm << 4, n0 = tn << 5;

    const int r  = lane & 15;
    const int kh = lane >> 4;
    const bf16_t* Ap = A + (size_t)(m0 + r) * K + (kh << 4);
    const bf16_t* Bp = Wt + (size_t)(kh << 3) * N + n0 + r;     // col n0+r, k base 8*kh
    const size_t  Bstep = (size_t)32 * N;
    auto ldB = [&](const bf16_t* p) { v16bf v; for (int s = 0; s < 16; ++s) v[s] = p[(size_t)(s + (s & 8)) * N]; return v; };

    // prologue: buffer 0 <- k-chunk 0
    v16bf a0 = *(const v16bf*)(Ap);
    v16bf p0 = ldB(Bp);
    v16bf q0 = ldB(Bp + 16);

    v8f c0 = {}, c1 = {};
    for (int k0 = 0; k0 < K; k0 += 64) {       // K % 64 == 0 (512/1024/2560)
        // buffer 1 <- k-chunk k0+32 (always valid)
        Ap += 32;
        Bp += Bstep;
        v16bf a1 = *(const v16bf*)(Ap);
        v16bf p1 = ldB(Bp);
        v16bf q1 = ldB(Bp + 16);
        __builtin_prefetch((const void*)(Bp + Bstep), 0, 1);
        c0 = WMMA_BF16(a0, p0, c0);
        c1 = WMMA_BF16(a0, q0, c1);
        if (k0 + 64 < K) {
            // buffer 0 <- k-chunk k0+64
            Ap += 32;
            Bp += Bstep;
            a0 = *(const v16bf*)(Ap);
            p0 = ldB(Bp);
            q0 = ldB(Bp + 16);
            __builtin_prefetch((const void*)(Bp + Bstep), 0, 1);
        }
        c0 = WMMA_BF16(a1, p1, c0);
        c1 = WMMA_BF16(a1, q1, c1);
    }

    const int na = n0 + r;
    const int nb = n0 + 16 + r;
    float bva = 0.0f, bvb = 0.0f;
    if (bias)  { bva += bias[na];  bvb += bias[nb];  }
    if (bias2) { bva += bias2[na]; bvb += bias2[nb]; }
    const int mbase = m0 + (kh << 3);
    const int sa = swz(na), sb = swz(nb);      // swizzled columns for bf16 copy
    if (Cbf) {
#pragma unroll
        for (int v = 0; v < 8; ++v) {
            float va = c0[v] + bva, vb = c1[v] + bvb;
            if (act == 1) { va = fmaxf(va, 0.0f); vb = fmaxf(vb, 0.0f); }
            const size_t rowoff = (size_t)(mbase + v) * N;
            C[rowoff + na] = va;           C[rowoff + nb] = vb;
            Cbf[rowoff + sa] = (bf16_t)va; Cbf[rowoff + sb] = (bf16_t)vb;
        }
    } else {
#pragma unroll
        for (int v = 0; v < 8; ++v) {
            float va = c0[v] + bva, vb = c1[v] + bvb;
            if (act == 1) { va = fmaxf(va, 0.0f); vb = fmaxf(vb, 0.0f); }
            const size_t rowoff = (size_t)(mbase + v) * N;
            C[rowoff + na] = va;
            C[rowoff + nb] = vb;
        }
    }
}

// ---------------- small helpers ----------------
// f32 -> bf16 with K-swizzled column layout (row width W)
__global__ void cvt_swz_bf16(const float* __restrict__ in, bf16_t* __restrict__ out,
                             size_t rows, int W)
{
    const size_t total = rows * W;
    for (size_t idx = blockIdx.x * (size_t)blockDim.x + threadIdx.x; idx < total;
         idx += (size_t)gridDim.x * blockDim.x) {
        size_t row = idx / W;
        int c = (int)(idx % W);
        out[row * W + swz(c)] = (bf16_t)in[idx];
    }
}
// in: (Nr, K) f32 row-major  ->  out: (K, Nr) bf16 row-major (K-major weight)
__global__ void cvt_transpose_bf16(const float* __restrict__ in, bf16_t* __restrict__ out,
                                   int Nr, int K)
{
    const size_t total = (size_t)Nr * K;
    for (size_t idx = blockIdx.x * (size_t)blockDim.x + threadIdx.x; idx < total;
         idx += (size_t)gridDim.x * blockDim.x) {
        size_t k = idx / Nr;
        int n = (int)(idx % Nr);
        out[idx] = (bf16_t)in[(size_t)n * K + k];
    }
}
__global__ void fill_f32(float* __restrict__ p, size_t n)
{
    for (size_t i = blockIdx.x * (size_t)blockDim.x + threadIdx.x; i < n;
         i += (size_t)gridDim.x * blockDim.x) p[i] = 0.0f;
}
__global__ void fill_bf16(bf16_t* __restrict__ p, size_t n)
{
    for (size_t i = blockIdx.x * (size_t)blockDim.x + threadIdx.x; i < n;
         i += (size_t)gridDim.x * blockDim.x) p[i] = (bf16_t)0.0f;
}
// one wave per row: out[row] = dot(H[row,:], w)
__global__ void rowdot_f32(const float* __restrict__ Hm, const float* __restrict__ w,
                           float* __restrict__ outv, int rows, int K)
{
    int row  = blockIdx.x * (blockDim.x >> 5) + (threadIdx.x >> 5);
    int lane = threadIdx.x & 31;
    if (row >= rows) return;
    const float* x = Hm + (size_t)row * K;
    float acc = 0.0f;
    for (int k = lane; k < K; k += 32) acc += x[k] * w[k];
    for (int o = 16; o; o >>= 1) acc += __shfl_down(acc, o, 32);
    if (lane == 0) outv[row] = acc;
}

// ---------------- GNN step kernels ----------------
__global__ void gnn_step0(const float* __restrict__ GI_C, const float* __restrict__ GH_P,
                          const float* __restrict__ cbhh, const float* __restrict__ pbih,
                          const float* __restrict__ Hl, float* __restrict__ Hn,
                          bf16_t* __restrict__ Hnbf, const float* __restrict__ wk,
                          float* __restrict__ K2)
{
    __shared__ float red[HIDD];
    const int b = blockIdx.x, d = threadIdx.x;
    const size_t row0 = (size_t)b * NN * GDIM;
    float r = sigmoidf_(GI_C[row0 + d] + cbhh[d]);
    float z = sigmoidf_(GI_C[row0 + HIDD + d] + cbhh[HIDD + d]);
    float nn = tanhf(GI_C[row0 + 2 * HIDD + d] + r * cbhh[2 * HIDD + d]);
    float C0 = (1.0f - z) * nn;
    float qr = sigmoidf_(pbih[d] + GH_P[row0 + d]);
    float qz = sigmoidf_(pbih[HIDD + d] + GH_P[row0 + HIDD + d]);
    float qn = tanhf(pbih[2 * HIDD + d] + qr * GH_P[row0 + 2 * HIDD + d]);
    float h0 = Hl[(size_t)b * NN * HIDD + d];
    float P0 = (1.0f - qz) * qn + qz * h0;
    float hv = C0 + P0;
    Hn[(size_t)b * NN * HIDD + d]        = hv;
    Hnbf[(size_t)b * NN * HIDD + swz(d)] = (bf16_t)hv;
    red[d] = hv * wk[d];
    __syncthreads();
    for (int s = HIDD / 2; s > 0; s >>= 1) { if (d < s) red[d] += red[d + s]; __syncthreads(); }
    if (d == 0) K2[b * NN] = red[0];
}

__global__ void gnn_attn(const float* __restrict__ Aq, const float* __restrict__ K2,
                         const float* __restrict__ adj, const float* __restrict__ H1,
                         const float* __restrict__ gatb, int l, int i,
                         float* __restrict__ Mf, bf16_t* __restrict__ Mbf)
{
    __shared__ float wgt[NN];
    __shared__ float red[256];
    const int b = blockIdx.x, t = threadIdx.x;  // 256 threads
    const float gb = gatb[l];
    float alpha = -3.0e38f;
    if (t < i) {
        float a = Aq[b * NN + i] + K2[b * NN + t] + gb;
        a -= (1.0f - adj[((size_t)b * NN + i) * NN + t]) * 1.0e30f;
        alpha = a;
    }
    red[t] = alpha;
    __syncthreads();
    for (int s = 128; s > 0; s >>= 1) { if (t < s) red[t] = fmaxf(red[t], red[t + s]); __syncthreads(); }
    float amax = red[0];
    __syncthreads();
    float e = (t < i) ? expf(alpha - amax) : 0.0f;
    red[t] = e;
    __syncthreads();
    for (int s = 128; s > 0; s >>= 1) { if (t < s) red[t] += red[t + s]; __syncthreads(); }
    float denom = red[0];
    if (t < NN) wgt[t] = (t < i) ? e / denom : 0.0f;
    __syncthreads();
    for (int d = t; d < HIDD; d += 256) {
        float acc = 0.0f;
        for (int n = 0; n < i; ++n)
            acc += wgt[n] * H1[((size_t)b * NN + n) * HIDD + d];
        Mf[b * HIDD + d]       = acc;
        Mbf[b * HIDD + swz(d)] = (bf16_t)acc;
    }
}

__global__ void gnn_gates(const float* __restrict__ GI_C, const float* __restrict__ GH_C,
                          const float* __restrict__ GI_P, const float* __restrict__ GH_P,
                          const float* __restrict__ Mf, const float* __restrict__ Hl,
                          float* __restrict__ Hn, bf16_t* __restrict__ Hnbf,
                          const float* __restrict__ wk, float* __restrict__ K2, int i)
{
    __shared__ float red[HIDD];
    const int b = blockIdx.x, d = threadIdx.x;
    const size_t rowi = ((size_t)b * NN + i) * GDIM;
    const size_t rowm = (size_t)b * GDIM;
    float Mv = Mf[b * HIDD + d];
    float hi = Hl[((size_t)b * NN + i) * HIDD + d];
    float r  = sigmoidf_(GI_C[rowi + d] + GH_C[rowm + d]);
    float z  = sigmoidf_(GI_C[rowi + HIDD + d] + GH_C[rowm + HIDD + d]);
    float nn = tanhf(GI_C[rowi + 2 * HIDD + d] + r * GH_C[rowm + 2 * HIDD + d]);
    float Cc = (1.0f - z) * nn + z * Mv;
    float qr = sigmoidf_(GI_P[rowm + d] + GH_P[rowi + d]);
    float qz = sigmoidf_(GI_P[rowm + HIDD + d] + GH_P[rowi + HIDD + d]);
    float qn = tanhf(GI_P[rowm + 2 * HIDD + d] + qr * GH_P[rowi + 2 * HIDD + d]);
    float Pp = (1.0f - qz) * qn + qz * hi;
    float hv = Cc + Pp;
    Hn[((size_t)b * NN + i) * HIDD + d]        = hv;
    Hnbf[((size_t)b * NN + i) * HIDD + swz(d)] = (bf16_t)hv;
    red[d] = hv * wk[d];
    __syncthreads();
    for (int s = HIDD / 2; s > 0; s >>= 1) { if (d < s) red[d] += red[d + s]; __syncthreads(); }
    if (d == 0) K2[b * NN + i] = red[0];
}

// ---------------- concat / mask / lstm / mlp ----------------
// pure physical copy: the K-swizzle commutes with concatenation because every
// block offset (0,512,1024,1536) is a multiple of 32.
__global__ void build_hcat(const bf16_t* __restrict__ H0, const bf16_t* __restrict__ H1,
                           const bf16_t* __restrict__ H2, const bf16_t* __restrict__ F,
                           bf16_t* __restrict__ Hcat)
{
    const size_t total = (size_t)ROWS * INDIM;
    for (size_t idx = blockIdx.x * (size_t)blockDim.x + threadIdx.x; idx < total;
         idx += (size_t)gridDim.x * blockDim.x) {
        size_t row = idx / INDIM;
        int col = (int)(idx % INDIM);
        bf16_t v;
        if      (col < HIDD)     v = H0[row * HIDD + col];
        else if (col < 2 * HIDD) v = H1[row * HIDD + col - HIDD];
        else if (col < 3 * HIDD) v = H2[row * HIDD + col - 2 * HIDD];
        else                     v = F[row * EMBD + col - 3 * HIDD];
        Hcat[idx] = v;
    }
}
__global__ void mask_split(const float* __restrict__ Hp, const int* __restrict__ t,
                           bf16_t* __restrict__ Abf, bf16_t* __restrict__ Bbf)
{
    const size_t total = (size_t)ROWS * HIDD;
    for (size_t idx = blockIdx.x * (size_t)blockDim.x + threadIdx.x; idx < total;
         idx += (size_t)gridDim.x * blockDim.x) {
        size_t row = idx / HIDD;
        int d = (int)(idx % HIDD);
        bool m = (t[row] == 1);
        float v = Hp[idx];
        Abf[row * HIDD + swz(d)] = (bf16_t)(m ? 0.0f : v);
        Bbf[row * HIDD + swz(d)] = (bf16_t)(m ? v : 0.0f);
    }
}
__global__ void lstm_gates(const float* __restrict__ X, const float* __restrict__ GH,
                           float* __restrict__ h, float* __restrict__ c,
                           bf16_t* __restrict__ hbf, float* __restrict__ mean, int n)
{
    const int b = blockIdx.x, d = threadIdx.x;
    const size_t rx = ((size_t)b * NN + n) * LDIM;
    const size_t rh = (size_t)b * LDIM;
    float gi = X[rx + d]            + GH[rh + d];
    float gf = X[rx + HIDD + d]     + GH[rh + HIDD + d];
    float gg = X[rx + 2 * HIDD + d] + GH[rh + 2 * HIDD + d];
    float go = X[rx + 3 * HIDD + d] + GH[rh + 3 * HIDD + d];
    float cv = sigmoidf_(gf) * c[b * HIDD + d] + sigmoidf_(gi) * tanhf(gg);
    float hv = sigmoidf_(go) * tanhf(cv);
    c[b * HIDD + d] = cv;
    h[b * HIDD + d] = hv;
    hbf[b * HIDD + swz(d)] = (bf16_t)hv;
    mean[b * HIDD + d] += hv * (1.0f / (float)NN);
}
__global__ void build_x0(const float* __restrict__ mA, const float* __restrict__ mB,
                         float* __restrict__ X0)
{
    const size_t total = (size_t)2 * BB * HIDD;
    for (size_t idx = blockIdx.x * (size_t)blockDim.x + threadIdx.x; idx < total;
         idx += (size_t)gridDim.x * blockDim.x) {
        size_t row = idx / HIDD;
        int d = (int)(idx % HIDD);
        size_t b = row >> 1;
        X0[idx] = (row & 1) ? mB[b * HIDD + d] : mA[b * HIDD + d];
    }
}
__global__ void fc_f32(const float* __restrict__ X, const float* __restrict__ W,
                       const float* __restrict__ b, float* __restrict__ Y,
                       int N, int K, int act)
{
    int row = blockIdx.x, col = threadIdx.x;
    if (col >= N) return;
    const float* x = X + (size_t)row * K;
    const float* w = W + (size_t)col * K;
    float acc = b[col];
    for (int k = 0; k < K; ++k) acc = fmaf(x[k], w[k], acc);
    if (act) acc = fmaxf(acc, 0.0f);
    Y[(size_t)row * N + col] = acc;
}

// ---------------------------------------------------------------------------
extern "C" void kernel_launch(void* const* d_in, const int* in_sizes, int n_in,
                              void* d_out, int out_size, void* d_ws, size_t ws_size,
                              hipStream_t stream)
{
    (void)in_sizes; (void)n_in; (void)out_size;
    const float* features = (const float*)d_in[0];
    const float* adj      = (const float*)d_in[1];
    const int*   tlab     = (const int*)  d_in[5];
    const float* fc1_w = (const float*)d_in[6];
    const float* fc1_b = (const float*)d_in[7];
    const float* gat_w = (const float*)d_in[8];
    const float* gat_b = (const float*)d_in[9];
    const float* gc_wih = (const float*)d_in[10];
    const float* gc_whh = (const float*)d_in[11];
    const float* gc_bih = (const float*)d_in[12];
    const float* gc_bhh = (const float*)d_in[13];
    const float* gp_wih = (const float*)d_in[14];
    const float* gp_whh = (const float*)d_in[15];
    const float* gp_bih = (const float*)d_in[16];
    const float* gp_bhh = (const float*)d_in[17];
    const float* pc_w = (const float*)d_in[18];
    const float* pc_b = (const float*)d_in[19];
    const float* wAih = (const float*)d_in[20];
    const float* wAhh = (const float*)d_in[21];
    const float* bAih = (const float*)d_in[22];
    const float* bAhh = (const float*)d_in[23];
    const float* wBih = (const float*)d_in[24];
    const float* wBhh = (const float*)d_in[25];
    const float* bBih = (const float*)d_in[26];
    const float* bBhh = (const float*)d_in[27];
    const float* m0w = (const float*)d_in[28];
    const float* m0b = (const float*)d_in[29];
    const float* m1w = (const float*)d_in[30];
    const float* m1b = (const float*)d_in[31];
    const float* m2w = (const float*)d_in[32];
    const float* m2b = (const float*)d_in[33];
    float* out = (float*)d_out;

    // ---- workspace bump allocator ----
    char* base = (char*)d_ws;
    size_t off = 0;
    auto alloc = [&](size_t bytes) -> void* {
        void* p = base + off;
        off = (off + bytes + 255) & ~(size_t)255;
        return p;
    };
    // K-major bf16 weights (B fragment elements read at stride N)
    bf16_t* fc1wt = (bf16_t*)alloc((size_t)HIDD * EMBD * 2);     // (EMBD, HIDD)
    bf16_t* gciht = (bf16_t*)alloc((size_t)2 * HIDD * GDIM * 2); // (HIDD, GDIM) per layer
    bf16_t* gchht = (bf16_t*)alloc((size_t)2 * HIDD * GDIM * 2);
    bf16_t* gpiht = (bf16_t*)alloc((size_t)2 * HIDD * GDIM * 2);
    bf16_t* gphht = (bf16_t*)alloc((size_t)2 * HIDD * GDIM * 2);
    bf16_t* pcwt  = (bf16_t*)alloc((size_t)INDIM * HIDD * 2);    // (INDIM, HIDD)
    bf16_t* aiht  = (bf16_t*)alloc((size_t)HIDD * LDIM * 2);     // (HIDD, LDIM)
    bf16_t* ahht  = (bf16_t*)alloc((size_t)HIDD * LDIM * 2);
    bf16_t* biht  = (bf16_t*)alloc((size_t)HIDD * LDIM * 2);
    bf16_t* bhht  = (bf16_t*)alloc((size_t)HIDD * LDIM * 2);
    bf16_t* feat_bf = (bf16_t*)alloc((size_t)ROWS * EMBD * 2);
    float*  Hf[3]; bf16_t* Hb[3];
    for (int l = 0; l < 3; ++l) {
        Hf[l] = (float*) alloc((size_t)ROWS * HIDD * 4);
        Hb[l] = (bf16_t*)alloc((size_t)ROWS * HIDD * 2);
    }
    float* big0 = (float*)alloc((size_t)ROWS * LDIM * 4);  // GI_C  | XA
    float* big1 = (float*)alloc((size_t)ROWS * LDIM * 4);  // GH_P  | XB
    float* GH_C = (float*)alloc((size_t)BB * GDIM * 4);
    float* GI_P = (float*)alloc((size_t)BB * GDIM * 4);
    float* GHA  = (float*)alloc((size_t)BB * LDIM * 4);
    float* GHB  = (float*)alloc((size_t)BB * LDIM * 4);
    bf16_t* hcat_bf = (bf16_t*)alloc((size_t)ROWS * INDIM * 2);
    float* Hp = (float*)alloc((size_t)ROWS * HIDD * 4);
    bf16_t* Abf = (bf16_t*)alloc((size_t)ROWS * HIDD * 2);
    bf16_t* Bbf = (bf16_t*)alloc((size_t)ROWS * HIDD * 2);
    float* Aq = (float*)alloc((size_t)ROWS * 4);
    float* K2 = (float*)alloc((size_t)ROWS * 4);
    float* Mf = (float*)alloc((size_t)BB * HIDD * 4);
    bf16_t* Mbf = (bf16_t*)alloc((size_t)BB * HIDD * 2);
    float* hA = (float*)alloc((size_t)BB * HIDD * 4);
    float* cA = (float*)alloc((size_t)BB * HIDD * 4);
    float* hB = (float*)alloc((size_t)BB * HIDD * 4);
    float* cB = (float*)alloc((size_t)BB * HIDD * 4);
    bf16_t* hAbf = (bf16_t*)alloc((size_t)BB * HIDD * 2);
    bf16_t* hBbf = (bf16_t*)alloc((size_t)BB * HIDD * 2);
    float* meanA = (float*)alloc((size_t)BB * HIDD * 4);
    float* meanB = (float*)alloc((size_t)BB * HIDD * 4);
    float* X0 = (float*)alloc((size_t)2 * BB * HIDD * 4);
    float* X1 = (float*)alloc((size_t)2 * BB * HIDD * 4);
    float* X2 = (float*)alloc((size_t)2 * BB * HIDD * 4);
    if (off > ws_size) return;  // deterministic no-op if workspace too small

    auto gsb = [](size_t n) { size_t b = (n + 255) / 256; return (int)(b > 4096 ? 4096 : b); };
    auto cvtS = [&](const float* s, bf16_t* d, size_t rows, int W) {
        cvt_swz_bf16<<<gsb(rows * W), 256, 0, stream>>>(s, d, rows, W);
    };
    auto cvtT = [&](const float* s, bf16_t* d, int Nr, int K) {
        size_t n = (size_t)Nr * K;
        cvt_transpose_bf16<<<gsb(n), 256, 0, stream>>>(s, d, Nr, K);
    };
    auto gemm = [&](const bf16_t* A, const bf16_t* Wt, const float* b1, const float* b2,
                    float* C, bf16_t* Cbf, int M, int Nc, int K, int act) {
        int tiles = (M / 16) * (Nc / 32);     // 16x32 macro-tiles
        gemm_bf16<<<(tiles + 7) / 8, 256, 0, stream>>>(A, Wt, b1, b2, C, Cbf, M, Nc, K, act);
    };
    auto zf = [&](float* p, size_t n)  { fill_f32 <<<gsb(n), 256, 0, stream>>>(p, n); };
    auto zb = [&](bf16_t* p, size_t n) { fill_bf16<<<gsb(n), 256, 0, stream>>>(p, n); };

    // ---- weight conversion to K-major bf16 ----
    cvtT(fc1_w, fc1wt, HIDD, EMBD);
    for (int l = 0; l < 2; ++l) {
        cvtT(gc_wih + (size_t)l * GDIM * HIDD, gciht + (size_t)l * HIDD * GDIM, GDIM, HIDD);
        cvtT(gc_whh + (size_t)l * GDIM * HIDD, gchht + (size_t)l * HIDD * GDIM, GDIM, HIDD);
        cvtT(gp_wih + (size_t)l * GDIM * HIDD, gpiht + (size_t)l * HIDD * GDIM, GDIM, HIDD);
        cvtT(gp_whh + (size_t)l * GDIM * HIDD, gphht + (size_t)l * HIDD * GDIM, GDIM, HIDD);
    }
    cvtT(pc_w, pcwt, HIDD, INDIM);
    cvtT(wAih, aiht, LDIM, HIDD);
    cvtT(wAhh, ahht, LDIM, HIDD);
    cvtT(wBih, biht, LDIM, HIDD);
    cvtT(wBhh, bhht, LDIM, HIDD);
    cvtS(features, feat_bf, ROWS, EMBD);   // K-swizzled activation copy

    // ---- stage 1: H0 = relu(features @ fc1_w.T + b) ----
    gemm(feat_bf, fc1wt, fc1_b, nullptr, Hf[0], Hb[0], ROWS, HIDD, EMBD, 1);

    // ---- stage 2: two GNN layers (input projections hoisted out of the scan) ----
    for (int l = 0; l < 2; ++l) {
        const float*  Hl   = Hf[l];
        const bf16_t* Hlbf = Hb[l];
        float*  Hn   = Hf[l + 1];
        bf16_t* Hnbf = Hb[l + 1];
        const bf16_t* cih = gciht + (size_t)l * HIDD * GDIM;
        const bf16_t* chh = gchht + (size_t)l * HIDD * GDIM;
        const bf16_t* pih = gpiht + (size_t)l * HIDD * GDIM;
        const bf16_t* phh = gphht + (size_t)l * HIDD * GDIM;
        const float* cbih = gc_bih + (size_t)l * GDIM;
        const float* cbhh = gc_bhh + (size_t)l * GDIM;
        const float* pbih = gp_bih + (size_t)l * GDIM;
        const float* pbhh = gp_bhh + (size_t)l * GDIM;
        const float* wq = gat_w + (size_t)l * 2 * HIDD;
        const float* wk = wq + HIDD;
        float* GI_C = big0;   // (ROWS, GDIM) = Hl @ Wc_ih.T + b_ih
        float* GH_P = big1;   // (ROWS, GDIM) = Hl @ Wp_hh.T + b_hh
        gemm(Hlbf, cih, cbih, nullptr, GI_C, nullptr, ROWS, GDIM, HIDD, 0);
        gemm(Hlbf, phh, pbhh, nullptr, GH_P, nullptr, ROWS, GDIM, HIDD, 0);
        rowdot_f32<<<ROWS / 8, 256, 0, stream>>>(Hl, wq, Aq, ROWS, HIDD);
        gnn_step0<<<BB, HIDD, 0, stream>>>(GI_C, GH_P, cbhh, pbih, Hl, Hn, Hnbf, wk, K2);
        for (int i = 1; i < NN; ++i) {
            gnn_attn<<<BB, 256, 0, stream>>>(Aq, K2, adj, Hn, gat_b, l, i, Mf, Mbf);
            gemm(Mbf, chh, cbhh, nullptr, GH_C, nullptr, BB, GDIM, HIDD, 0);
            gemm(Mbf, pih, pbih, nullptr, GI_P, nullptr, BB, GDIM, HIDD, 0);
            gnn_gates<<<BB, HIDD, 0, stream>>>(GI_C, GH_C, GI_P, GH_P, Mf, Hl, Hn, Hnbf, wk, K2, i);
        }
    }

    // ---- stage 3: Hcat -> Hp ----
    build_hcat<<<4096, 256, 0, stream>>>(Hb[0], Hb[1], Hb[2], feat_bf, hcat_bf);
    gemm(hcat_bf, pcwt, pc_b, nullptr, Hp, nullptr, ROWS, HIDD, INDIM, 0);

    // ---- stage 4: mask split, LSTM input projections, recurrence ----
    mask_split<<<4096, 256, 0, stream>>>(Hp, tlab, Abf, Bbf);
    float* XA = big0;
    float* XB = big1;
    gemm(Abf, aiht, bAih, bAhh, XA, nullptr, ROWS, LDIM, HIDD, 0);
    gemm(Bbf, biht, bBih, bBhh, XB, nullptr, ROWS, LDIM, HIDD, 0);
    zf(hA, (size_t)BB * HIDD); zf(cA, (size_t)BB * HIDD);
    zf(hB, (size_t)BB * HIDD); zf(cB, (size_t)BB * HIDD);
    zf(meanA, (size_t)BB * HIDD); zf(meanB, (size_t)BB * HIDD);
    zb(hAbf, (size_t)BB * HIDD); zb(hBbf, (size_t)BB * HIDD);
    for (int n = 0; n < NN; ++n) {
        gemm(hAbf, ahht, nullptr, nullptr, GHA, nullptr, BB, LDIM, HIDD, 0);
        gemm(hBbf, bhht, nullptr, nullptr, GHB, nullptr, BB, LDIM, HIDD, 0);
        lstm_gates<<<BB, HIDD, 0, stream>>>(XA, GHA, hA, cA, hAbf, meanA, n);
        lstm_gates<<<BB, HIDD, 0, stream>>>(XB, GHB, hB, cB, hBbf, meanB, n);
    }

    // ---- stage 5: MLP head (fp32 exact) ----
    build_x0<<<256, 256, 0, stream>>>(meanA, meanB, X0);
    fc_f32<<<2 * BB, 512, 0, stream>>>(X0, m0w, m0b, X1, HIDD, HIDD, 1);
    fc_f32<<<2 * BB, 512, 0, stream>>>(X1, m1w, m1b, X2, HIDD, HIDD, 1);
    fc_f32<<<2 * BB, 64, 0, stream>>>(X2, m2w, m2b, out, NCLS, HIDD, 0);
}
